// GNP_20615843020982
// MI455X (gfx1250) — matrix operand order repaired and checked by reading, not compiled, not run on hardware
//
#include <hip/hip_runtime.h>
#include <math.h>

#define Bq 8
#define Nq 2048
#define Hq 64
#define Dq 16
#define Lq 8
#define EPSq 1e-8f

typedef __attribute__((ext_vector_type(16))) __bf16 v16bf;
typedef __attribute__((ext_vector_type(8)))  __bf16 v8bf;
typedef __attribute__((ext_vector_type(8)))  float  v8f;

// ---------------- tau / scale per batch ----------------
__global__ void tau_kernel(const float* __restrict__ b, const int* __restrict__ sizes,
                           float* __restrict__ scale_in, float* __restrict__ scale_out) {
  __shared__ float red[256];
  const int bb = blockIdx.x, t = threadIdx.x;
  const int n = sizes[bb];
  const float* bp = b + (size_t)bb * Nq;
  float s = 0.f;
  for (int i = t; i < Nq; i += 256) {
    float v = (i < n) ? bp[i] : 0.f;
    s += v * v;
  }
  red[t] = s; __syncthreads();
  for (int off = 128; off > 0; off >>= 1) {
    if (t < off) red[t] += red[t + off];
    __syncthreads();
  }
  if (t == 0) {
    float tau = sqrtf(red[0]) + EPSq;
    float sq  = sqrtf((float)n);
    scale_in[bb]  = sq / tau;
    scale_out[bb] = tau / sq;
  }
}

// ---------------- one-time A_hat f32 -> bf16 conversion ----------------
__global__ void convA_kernel(const float* __restrict__ A, __bf16* __restrict__ Abf) {
  const size_t idx = ((size_t)blockIdx.x * 256 + threadIdx.x) * 8;
  float4 a0 = *(const float4*)(A + idx);
  float4 a1 = *(const float4*)(A + idx + 4);
  v8bf o;
  o[0] = (__bf16)a0.x; o[1] = (__bf16)a0.y; o[2] = (__bf16)a0.z; o[3] = (__bf16)a0.w;
  o[4] = (__bf16)a1.x; o[5] = (__bf16)a1.y; o[6] = (__bf16)a1.z; o[7] = (__bf16)a1.w;
  *(v8bf*)(Abf + idx) = o;
}

// ---------------- encoder MLP: scalar -> H -> D ----------------
__global__ void encoder_kernel(const float* __restrict__ b, const int* __restrict__ sizes,
                               const float* __restrict__ scale_in,
                               const float* __restrict__ W1, const float* __restrict__ b1,
                               const float* __restrict__ W2, const float* __restrict__ b2,
                               float* __restrict__ X) {
  __shared__ float sW1[Hq], sb1[Hq], sW2[Hq * Dq], sb2[Dq];
  const int t = threadIdx.x;
  if (t < Hq) { sW1[t] = W1[t]; sb1[t] = b1[t]; }
  if (t < Dq) sb2[t] = b2[t];
  for (int i = t; i < Hq * Dq; i += 256) sW2[i] = W2[i];
  __syncthreads();
  const int gid = blockIdx.x * 256 + t;      // node id in [0, B*N)
  const int bb = gid / Nq, i = gid % Nq;
  const int n = sizes[bb];
  const float bs = ((i < n) ? b[gid] : 0.f) * scale_in[bb];
  float acc[Dq];
  #pragma unroll
  for (int d = 0; d < Dq; d++) acc[d] = sb2[d];
  for (int h = 0; h < Hq; h++) {
    float hv = fmaxf(bs * sW1[h] + sb1[h], 0.f);
    #pragma unroll
    for (int d = 0; d < Dq; d++) acc[d] += hv * sW2[h * Dq + d];
  }
  float* xp = X + (size_t)gid * Dq;
  #pragma unroll
  for (int d = 0; d < Dq; d++) xp[d] = fmaxf(acc[d], 0.f);
}

// ---------------- per-layer: Y^T = (X*Wg)^T (bf16, col-masked), XU = X*U (f32) ----------------
__global__ void xw_kernel(const float* __restrict__ X,
                          const float* __restrict__ Wg_l, const float* __restrict__ U_l,
                          const int* __restrict__ sizes,
                          __bf16* __restrict__ YT, float* __restrict__ XU) {
  __shared__ float sW[Dq * Dq], sU[Dq * Dq];
  const int t = threadIdx.x;
  sW[t] = Wg_l[t];
  sU[t] = U_l[t];
  __syncthreads();
  const int gid = blockIdx.x * 256 + t;
  const int bb = gid / Nq, j = gid % Nq;
  const int n = sizes[bb];
  float x[Dq];
  const float* xp = X + (size_t)gid * Dq;
  #pragma unroll
  for (int d = 0; d < Dq; d++) x[d] = xp[d];
  float* xup = XU + (size_t)gid * Dq;
  __bf16* ytp = YT + (size_t)bb * Dq * Nq + j;     // [b][e][j], stride N per e
  const bool valid = (j < n);
  #pragma unroll
  for (int e = 0; e < Dq; e++) {
    float yw = 0.f, xu = 0.f;
    #pragma unroll
    for (int d = 0; d < Dq; d++) {
      yw += x[d] * sW[d * Dq + e];
      xu += x[d] * sU[d * Dq + e];
    }
    xup[e] = xu;
    ytp[(size_t)e * Nq] = valid ? (__bf16)yw : (__bf16)0.f;
  }
}

// ---------------- shared epilogue for both GEMM variants ----------------
__device__ __forceinline__ void gemm_epilogue(v8f acc, int bb, int tile, int half, int l16,
                                              int n, const float* __restrict__ XU,
                                              float* __restrict__ Xn) {
  const float* xup = XU + ((size_t)bb * Nq + tile) * Dq + l16;
  float*       xnp = Xn + ((size_t)bb * Nq + tile) * Dq + l16;
  #pragma unroll
  for (int r = 0; r < 8; r++) {
    const int m = r + half * 8;
    float a = ((tile + m) < n) ? acc[r] : 0.f;   // row mask on A
    float v = fmaxf(a + xup[(size_t)m * Dq], 0.f);
    xnp[(size_t)m * Dq] = v;
  }
}

// ---------------- per-layer big GEMM (bf16 pre-converted A) ----------------
__global__ void __launch_bounds__(256) gcn_wmma_bf16_kernel(const __bf16* __restrict__ Abf,
                                                            const __bf16* __restrict__ YT,
                                                            const float* __restrict__ XU,
                                                            const int* __restrict__ sizes,
                                                            float* __restrict__ Xn) {
  const int t    = threadIdx.x;
  const int wave = t >> 5;
  const int lane = t & 31;
  const int half = lane >> 4;
  const int l16  = lane & 15;

  const int bb   = blockIdx.x >> 4;
  const int tile = ((blockIdx.x & 15) * 8 + wave) * 16;
  const int n    = sizes[bb];

  const __bf16* Arow = Abf + ((size_t)bb * Nq + (tile + l16)) * Nq;
  const __bf16* Brow = YT  + ((size_t)bb * Dq + l16) * Nq;
  const int cA = half * 16;
  const int cB = half * 16;

  v8f acc = {};
  #pragma unroll 8
  for (int k0 = 0; k0 < Nq; k0 += 32) {
    v8bf alo = *(const v8bf*)(Arow + k0 + cA);         // K = c..c+7
    v8bf ahi = *(const v8bf*)(Arow + k0 + cA + 8);     // K = c+8..c+15
    v16bf af = __builtin_shufflevector(alo, ahi, 0, 1, 2, 3, 4, 5, 6, 7,
                                       8, 9, 10, 11, 12, 13, 14, 15);
    v16bf bfv = *(const v16bf*)(Brow + k0 + cB);
    acc = __builtin_amdgcn_wmma_f32_16x16x32_bf16(false, af, false, bfv,
                                                  (short)0, acc, false, false);
  }
  gemm_epilogue(acc, bb, tile, half, l16, n, XU, Xn);
}

// ---------------- per-layer big GEMM (inline f32->bf16 convert; ws fallback) ----------------
__global__ void __launch_bounds__(256) gcn_wmma_kernel(const float* __restrict__ A,
                                                       const __bf16* __restrict__ YT,
                                                       const float* __restrict__ XU,
                                                       const int* __restrict__ sizes,
                                                       float* __restrict__ Xn) {
  const int t    = threadIdx.x;
  const int wave = t >> 5;
  const int lane = t & 31;
  const int half = lane >> 4;
  const int l16  = lane & 15;

  const int bb   = blockIdx.x >> 4;
  const int tile = ((blockIdx.x & 15) * 8 + wave) * 16;
  const int n    = sizes[bb];

  const float*  Arow = A  + ((size_t)bb * Nq + (tile + l16)) * Nq;
  const __bf16* Brow = YT + ((size_t)bb * Dq + l16) * Nq;
  const int cA = half * 16;
  const int cB = half * 16;

  v8f acc = {};
  #pragma unroll 4
  for (int k0 = 0; k0 < Nq; k0 += 32) {
    const float* ap = Arow + k0 + cA;
    float4 a0 = *(const float4*)(ap);
    float4 a1 = *(const float4*)(ap + 4);
    float4 a2 = *(const float4*)(ap + 8);
    float4 a3 = *(const float4*)(ap + 12);
    v16bf af;
    af[0]  = (__bf16)a0.x; af[1]  = (__bf16)a0.y; af[2]  = (__bf16)a0.z; af[3]  = (__bf16)a0.w;
    af[4]  = (__bf16)a1.x; af[5]  = (__bf16)a1.y; af[6]  = (__bf16)a1.z; af[7]  = (__bf16)a1.w;
    af[8]  = (__bf16)a2.x; af[9]  = (__bf16)a2.y; af[10] = (__bf16)a2.z; af[11] = (__bf16)a2.w;
    af[12] = (__bf16)a3.x; af[13] = (__bf16)a3.y; af[14] = (__bf16)a3.z; af[15] = (__bf16)a3.w;

    v16bf bfv = *(const v16bf*)(Brow + k0 + cB);
    acc = __builtin_amdgcn_wmma_f32_16x16x32_bf16(false, af, false, bfv,
                                                  (short)0, acc, false, false);
  }
  gemm_epilogue(acc, bb, tile, half, l16, n, XU, Xn);
}

// ---------------- decoder MLP: D -> H -> 1, scale + mask ----------------
__global__ void decoder_kernel(const float* __restrict__ X, const int* __restrict__ sizes,
                               const float* __restrict__ scale_out,
                               const float* __restrict__ W1, const float* __restrict__ b1,
                               const float* __restrict__ W2, const float* __restrict__ b2,
                               float* __restrict__ out) {
  __shared__ float sW1[Dq * Hq], sb1[Hq], sW2[Hq];
  const int t = threadIdx.x;
  for (int i = t; i < Dq * Hq; i += 256) sW1[i] = W1[i];
  if (t < Hq) { sb1[t] = b1[t]; sW2[t] = W2[t]; }
  __syncthreads();
  const int gid = blockIdx.x * 256 + t;
  const int bb = gid / Nq, i = gid % Nq;
  const int n = sizes[bb];
  float x[Dq];
  const float* xp = X + (size_t)gid * Dq;
  #pragma unroll
  for (int d = 0; d < Dq; d++) x[d] = xp[d];
  float acc = b2[0];
  for (int h = 0; h < Hq; h++) {
    float hv = sb1[h];
    #pragma unroll
    for (int d = 0; d < Dq; d++) hv += x[d] * sW1[d * Hq + h];
    acc += fmaxf(hv, 0.f) * sW2[h];
  }
  out[gid] = (i < n) ? acc * scale_out[bb] : 0.f;
}

extern "C" void kernel_launch(void* const* d_in, const int* in_sizes, int n_in,
                              void* d_out, int out_size, void* d_ws, size_t ws_size,
                              hipStream_t stream) {
  (void)in_sizes; (void)n_in; (void)out_size;
  const float* b     = (const float*)d_in[0];
  const float* A     = (const float*)d_in[1];
  const int*   sizes = (const int*)  d_in[2];
  const float* eW1   = (const float*)d_in[3];
  const float* eb1   = (const float*)d_in[4];
  const float* eW2   = (const float*)d_in[5];
  const float* eb2   = (const float*)d_in[6];
  const float* U     = (const float*)d_in[7];
  const float* Wg    = (const float*)d_in[8];
  const float* dW1   = (const float*)d_in[9];
  const float* db1   = (const float*)d_in[10];
  const float* dW2   = (const float*)d_in[11];
  const float* db2   = (const float*)d_in[12];

  char* ws = (char*)d_ws;
  size_t off = 0;
  auto alloc = [&](size_t bytes) -> char* {
    char* p = ws + off;
    off += (bytes + 255) & ~(size_t)255;
    return p;
  };
  // small buffers first so the fallback path always fits
  float*  scale_in  = (float*) alloc(Bq * sizeof(float));
  float*  scale_out = (float*) alloc(Bq * sizeof(float));
  float*  X0        = (float*) alloc((size_t)Bq * Nq * Dq * sizeof(float));
  float*  X1        = (float*) alloc((size_t)Bq * Nq * Dq * sizeof(float));
  __bf16* YT        = (__bf16*)alloc((size_t)Bq * Dq * Nq * sizeof(__bf16));
  float*  XU        = (float*) alloc((size_t)Bq * Nq * Dq * sizeof(float));
  const size_t abf_bytes = (size_t)Bq * Nq * Nq * sizeof(__bf16);   // 64 MB
  __bf16* Abf       = (__bf16*)alloc(abf_bytes);
  const bool use_bf16_A = (off <= ws_size);   // ws_size fixed per run -> deterministic path

  tau_kernel<<<Bq, 256, 0, stream>>>(b, sizes, scale_in, scale_out);
  encoder_kernel<<<(Bq * Nq) / 256, 256, 0, stream>>>(b, sizes, scale_in,
                                                      eW1, eb1, eW2, eb2, X0);
  if (use_bf16_A) {
    convA_kernel<<<(int)(((size_t)Bq * Nq * Nq / 8) / 256), 256, 0, stream>>>(A, Abf);
  }

  float* Xc = X0;
  float* Xn = X1;
  for (int l = 0; l < Lq; l++) {
    xw_kernel<<<(Bq * Nq) / 256, 256, 0, stream>>>(Xc, Wg + l * Dq * Dq, U + l * Dq * Dq,
                                                   sizes, YT, XU);
    if (use_bf16_A) {
      gcn_wmma_bf16_kernel<<<Bq * (Nq / 16) / 8, 256, 0, stream>>>(Abf, YT, XU, sizes, Xn);
    } else {
      gcn_wmma_kernel<<<Bq * (Nq / 16) / 8, 256, 0, stream>>>(A, YT, XU, sizes, Xn);
    }
    float* tmp = Xc; Xc = Xn; Xn = tmp;
  }
  decoder_kernel<<<(Bq * Nq) / 256, 256, 0, stream>>>(Xc, sizes, scale_out,
                                                      dW1, db1, dW2, db2, (float*)d_out);
}
